// Decoder_27977416966319
// MI455X (gfx1250) — matrix operand order repaired and checked by reading, not compiled, not run on hardware
//
#include <hip/hip_runtime.h>
#include <hip/hip_bf16.h>
#include <math.h>

// ---------------------------------------------------------------------------
// Problem constants
// ---------------------------------------------------------------------------
#define BDIM 128
#define SDIM 1000
#define HDIM 256
#define MT32 ((SDIM + 31) / 32)     // 32  (32-row s-tiles for WMMA kernels)

typedef __attribute__((ext_vector_type(16))) __bf16 v16bf;
typedef __attribute__((ext_vector_type(8)))  float  v8f;

// Branch-free tanh: 1 - 2/(exp(2x)+1); correct limits at +-inf, no exec mask.
__device__ __forceinline__ float tanh_fast(float x) {
    return 1.0f - 2.0f / (__expf(2.0f * x) + 1.0f);
}

// ---------------------------------------------------------------------------
// WMMA fragment layouts (bf16 16x16x32, wave32, CDNA5 ISA 7.12.2)
// A 16x32 (MxK): lane<16 -> row M=lane, K = kk+[0..7] and kk+[16..23]
//                lane>=16 -> row M=lane-16, K = kk+[8..15] and kk+[24..31]
// B 32x16 (KxN): lane L -> col N = L&15, K same as A (kk+(L>>4)*8+[0..7], +16)
// ---------------------------------------------------------------------------
__device__ __forceinline__ v16bf load_a_frag_f32(const float* __restrict__ row,
                                                 int kk, int half) {
    const float4* p0 = (const float4*)(row + kk + half * 8);
    const float4* p1 = (const float4*)(row + kk + 16 + half * 8);
    float4 f0 = p0[0], f1 = p0[1];
    float4 f2 = p1[0], f3 = p1[1];
    v16bf a;
    a[0]  = (__bf16)f0.x; a[1]  = (__bf16)f0.y; a[2]  = (__bf16)f0.z; a[3]  = (__bf16)f0.w;
    a[4]  = (__bf16)f1.x; a[5]  = (__bf16)f1.y; a[6]  = (__bf16)f1.z; a[7]  = (__bf16)f1.w;
    a[8]  = (__bf16)f2.x; a[9]  = (__bf16)f2.y; a[10] = (__bf16)f2.z; a[11] = (__bf16)f2.w;
    a[12] = (__bf16)f3.x; a[13] = (__bf16)f3.y; a[14] = (__bf16)f3.z; a[15] = (__bf16)f3.w;
    return a;
}

// ---------------------------------------------------------------------------
// Kernel 0: pack a weight matrix slice W[:, 0:ksteps*32] (row-major, ld=ldw)
// into bf16 WMMA B-fragments: Wp[((nt*ksteps + ks)*32 + lane)*16 + i].
// ---------------------------------------------------------------------------
__global__ __launch_bounds__(256) void pack_w_kernel(
    const float* __restrict__ W, __bf16* __restrict__ Wp,
    int ldw, int ksteps)
{
    const int flat    = blockIdx.x * 256 + threadIdx.x;
    const int lane    = flat & 31;
    const int tileIdx = flat >> 5;            // nt*ksteps + ks
    const int ks      = tileIdx % ksteps;
    const int nt      = tileIdx / ksteps;
    const int col     = nt * 16 + (lane & 15);
    const int kbase   = ks * 32 + (lane >> 4) * 8;
    const float* src  = W + (size_t)col * ldw + kbase;
    v16bf v;
    #pragma unroll
    for (int i = 0; i < 16; ++i) v[i] = (__bf16)src[i < 8 ? i : i + 8];
    *(v16bf*)(Wp + ((size_t)flat << 4)) = v;
}

// ---------------------------------------------------------------------------
// Kernel 1: embedding + GRU cell. One block per batch row, 256 threads (h).
// ---------------------------------------------------------------------------
__global__ __launch_bounds__(256) void gru_kernel(
    const float* __restrict__ di,  const float* __restrict__ lh,
    const float* __restrict__ embW, const float* __restrict__ embb,
    const float* __restrict__ Wih, const float* __restrict__ Whh,
    const float* __restrict__ bih, const float* __restrict__ bhh,
    float* __restrict__ h_new, float* __restrict__ hidden_out)
{
    const int b = blockIdx.x;
    const int h = threadIdx.x;
    __shared__ float emb_s[HDIM];
    __shared__ float hs[HDIM];

    const float d0 = di[b * 2 + 0];
    const float d1 = di[b * 2 + 1];
    emb_s[h] = embb[h] + d0 * embW[h * 2 + 0] + d1 * embW[h * 2 + 1];
    hs[h]    = lh[(size_t)b * HDIM + h];
    __syncthreads();

    float xr = bih[h], xz = bih[HDIM + h], xn = bih[2 * HDIM + h];
    float hr = bhh[h], hz = bhh[HDIM + h], hn = bhh[2 * HDIM + h];
    const float* wr = Wih + (size_t)h * HDIM;
    const float* wz = Wih + (size_t)(HDIM + h) * HDIM;
    const float* wn = Wih + (size_t)(2 * HDIM + h) * HDIM;
    const float* ur = Whh + (size_t)h * HDIM;
    const float* uz = Whh + (size_t)(HDIM + h) * HDIM;
    const float* un = Whh + (size_t)(2 * HDIM + h) * HDIM;

    #pragma unroll 4
    for (int k = 0; k < HDIM; ++k) {
        const float ek = emb_s[k];
        const float hk = hs[k];
        xr += wr[k] * ek;  xz += wz[k] * ek;  xn += wn[k] * ek;
        hr += ur[k] * hk;  hz += uz[k] * hk;  hn += un[k] * hk;
    }
    const float r = 1.0f / (1.0f + __expf(-(xr + hr)));
    const float z = 1.0f / (1.0f + __expf(-(xz + hz)));
    const float n = tanh_fast(xn + r * hn);
    const float out = (1.0f - z) * n + z * hs[h];
    h_new[(size_t)b * HDIM + h]      = out;
    hidden_out[(size_t)b * HDIM + h] = out;
}

// ---------------------------------------------------------------------------
// Kernel 2: Ah[b,j] = sum_k attn_W[j, 2H+k] * h_new[b,k]   (j in [0,768))
// ---------------------------------------------------------------------------
__global__ __launch_bounds__(256) void ah_kernel(
    const float* __restrict__ attnW, const float* __restrict__ h_new,
    float* __restrict__ Ah)
{
    const int b = blockIdx.x;
    const int j = blockIdx.y * 256 + threadIdx.x;
    __shared__ float hs[HDIM];
    hs[threadIdx.x] = h_new[(size_t)b * HDIM + threadIdx.x];
    __syncthreads();
    const float* w = attnW + (size_t)j * (3 * HDIM) + 2 * HDIM;
    float acc = 0.0f;
    #pragma unroll 4
    for (int k = 0; k < HDIM; ++k) acc += w[k] * hs[k];
    Ah[(size_t)b * (3 * HDIM) + j] = acc;
}

// ---------------------------------------------------------------------------
// Kernel 3: attention raw scores via WMMA.
// scores[b,s] = attn_v . tanh( W1@static[b,s] + W2@dynamic[b,s] + Ah[b] )
// One block per (b, 32-row s-tile): two M-tiles share every B fragment, so
// each B load feeds two back-to-back WMMAs (independent accumulators).
// 8 waves x 6 N-tiles cover N=768; K = 512 in 16 steps of 32.
// ---------------------------------------------------------------------------
__global__ __launch_bounds__(256) void attn_scores_kernel(
    const float* __restrict__ statc, const float* __restrict__ dynamc,
    const __bf16* __restrict__ WaP, const float* __restrict__ attnv,
    const float* __restrict__ Ah, float* __restrict__ scores)
{
    const int b    = blockIdx.x / MT32;
    const int st   = blockIdx.x % MT32;
    const int s0   = st * 32;
    const int tid  = threadIdx.x;
    const int wave = tid >> 5;
    const int lane = tid & 31;
    const int half = lane >> 4;
    const int mrow = lane & 15;

    __shared__ __align__(16) __bf16 apanel[16 * 2 * 32 * 16];   // 32KB
    __shared__ float sm[32];

    // Stage A panel: slots = (ks 0..15) x (mtile 0..1) x (lane 0..31)
    // ks<8 -> static, ks>=8 -> dynamic.
    if (tid < 32) sm[tid] = 0.0f;
    #pragma unroll
    for (int slot = tid; slot < 1024; slot += 256) {
        const int ksl = slot >> 6;
        const int mt  = (slot >> 5) & 1;
        const int ln  = slot & 31;
        const int hf  = ln >> 4;
        int sr = s0 + mt * 16 + (ln & 15);
        if (sr >= SDIM) sr = SDIM - 1;      // clamp; masked on store
        const float* row = (ksl < 8 ? statc : dynamc)
                           + ((size_t)b * SDIM + sr) * HDIM;
        const v16bf v = load_a_frag_f32(row, (ksl & 7) * 32, hf);
        *(v16bf*)(apanel + (slot << 4)) = v;
    }
    __syncthreads();

    const __bf16* bbase = WaP + ((size_t)((wave * 6) * 16 * 32 + lane) << 4);
    const __bf16* abase = apanel + (lane << 4);

    v8f acc[6][2] = {};
    #pragma unroll 2
    for (int ks = 0; ks < 16; ++ks) {
        const v16bf a0 = *(const v16bf*)(abase + (ks * 2 + 0) * (32 * 16));
        const v16bf a1 = *(const v16bf*)(abase + (ks * 2 + 1) * (32 * 16));
        #pragma unroll
        for (int t = 0; t < 6; ++t) {
            const v16bf bf =
                *(const v16bf*)(bbase + (size_t)(t * 16 + ks) * (32 * 16));
            acc[t][0] = __builtin_amdgcn_wmma_f32_16x16x32_bf16(
                false, a0, false, bf, (short)0, acc[t][0], false, false);
            acc[t][1] = __builtin_amdgcn_wmma_f32_16x16x32_bf16(
                false, a1, false, bf, (short)0, acc[t][1], false, false);
        }
    }

    // Epilogue: tanh, dot with attn_v, reduce rows across lanes/waves.
    const int ncol = wave * 96 + mrow;
    float bias[6], vv[6];
    #pragma unroll
    for (int t = 0; t < 6; ++t) {
        const int n = ncol + t * 16;
        bias[t] = Ah[(size_t)b * (3 * HDIM) + n];
        vv[t]   = attnv[n];
    }
    #pragma unroll
    for (int r = 0; r < 8; ++r) {   // acc VGPR r -> row mt*16 + r + 8*half
        float x0 = 0.0f, x1 = 0.0f;
        #pragma unroll
        for (int t = 0; t < 6; ++t) {
            x0 += tanh_fast(acc[t][0][r] + bias[t]) * vv[t];
            x1 += tanh_fast(acc[t][1][r] + bias[t]) * vv[t];
        }
        x0 += __shfl_xor(x0, 1); x1 += __shfl_xor(x1, 1);
        x0 += __shfl_xor(x0, 2); x1 += __shfl_xor(x1, 2);
        x0 += __shfl_xor(x0, 4); x1 += __shfl_xor(x1, 4);
        x0 += __shfl_xor(x0, 8); x1 += __shfl_xor(x1, 8);
        if (mrow == r) {
            atomicAdd(&sm[r + half * 8], x0);
            atomicAdd(&sm[16 + r + half * 8], x1);
        }
    }
    __syncthreads();
    if (tid < 32 && (s0 + tid) < SDIM)
        scores[(size_t)b * SDIM + s0 + tid] = sm[tid];
}

// ---------------------------------------------------------------------------
// Kernel 4: softmax over S + context = attn @ static. One block per b.
// ---------------------------------------------------------------------------
__global__ __launch_bounds__(256) void softmax_context_kernel(
    const float* __restrict__ scores, const float* __restrict__ statc,
    float* __restrict__ context)
{
    const int b = blockIdx.x;
    const int tid = threadIdx.x;
    __shared__ float att[SDIM];
    __shared__ float red[256];

    float mx = -3.0e38f;
    for (int s = tid; s < SDIM; s += 256) {
        const float v = scores[(size_t)b * SDIM + s];
        att[s] = v;
        mx = fmaxf(mx, v);
    }
    red[tid] = mx;
    __syncthreads();
    for (int off = 128; off > 0; off >>= 1) {
        if (tid < off) red[tid] = fmaxf(red[tid], red[tid + off]);
        __syncthreads();
    }
    mx = red[0];
    __syncthreads();

    float sum = 0.0f;
    for (int s = tid; s < SDIM; s += 256) {
        const float e = __expf(att[s] - mx);
        att[s] = e;
        sum += e;
    }
    red[tid] = sum;
    __syncthreads();
    for (int off = 128; off > 0; off >>= 1) {
        if (tid < off) red[tid] += red[tid + off];
        __syncthreads();
    }
    const float inv = 1.0f / red[0];
    __syncthreads();

    // context[b, h=tid] = sum_s att[s] * static[b,s,h] * inv  (coalesced in h)
    float c = 0.0f;
    const float* sp = statc + (size_t)b * SDIM * HDIM + tid;
    #pragma unroll 4
    for (int s = 0; s < SDIM; ++s) c += att[s] * sp[(size_t)s * HDIM];
    context[(size_t)b * HDIM + tid] = c * inv;
}

// ---------------------------------------------------------------------------
// Kernel 5: Dc[b,j] = sum_k dec_W[j, H+k] * context[b,k]  (j in [0,512))
// ---------------------------------------------------------------------------
__global__ __launch_bounds__(256) void dc_kernel(
    const float* __restrict__ decW, const float* __restrict__ context,
    float* __restrict__ Dc)
{
    const int b = blockIdx.x;
    const int j = blockIdx.y * 256 + threadIdx.x;
    __shared__ float cs[HDIM];
    cs[threadIdx.x] = context[(size_t)b * HDIM + threadIdx.x];
    __syncthreads();
    const float* w = decW + (size_t)j * (2 * HDIM) + HDIM;
    float acc = 0.0f;
    #pragma unroll 4
    for (int k = 0; k < HDIM; ++k) acc += w[k] * cs[k];
    Dc[(size_t)b * (2 * HDIM) + j] = acc;
}

// ---------------------------------------------------------------------------
// Kernel 6: pointer raw scores via WMMA.
// scores2[b,s] = dec_v . tanh( Ws@static[b,s] + Dc[b] )
// One block per (b, 32-row s-tile); 8 waves x 4 N-tiles cover N=512;
// K = 256 in 8 steps; every B fragment feeds two WMMAs.
// ---------------------------------------------------------------------------
__global__ __launch_bounds__(256) void ptr_scores_kernel(
    const float* __restrict__ statc, const __bf16* __restrict__ WdP,
    const float* __restrict__ decv, const float* __restrict__ Dc,
    float* __restrict__ scores2)
{
    const int b    = blockIdx.x / MT32;
    const int st   = blockIdx.x % MT32;
    const int s0   = st * 32;
    const int tid  = threadIdx.x;
    const int wave = tid >> 5;
    const int lane = tid & 31;
    const int half = lane >> 4;
    const int mrow = lane & 15;

    __shared__ __align__(16) __bf16 apanel[8 * 2 * 32 * 16];    // 16KB
    __shared__ float sm[32];

    // Stage A panel: slots = (ks 0..7) x (mtile 0..1) x (lane 0..31); 2/thread
    if (tid < 32) sm[tid] = 0.0f;
    #pragma unroll
    for (int slot = tid; slot < 512; slot += 256) {
        const int ksl = slot >> 6;
        const int mt  = (slot >> 5) & 1;
        const int ln  = slot & 31;
        const int hf  = ln >> 4;
        int sr = s0 + mt * 16 + (ln & 15);
        if (sr >= SDIM) sr = SDIM - 1;
        const float* row = statc + ((size_t)b * SDIM + sr) * HDIM;
        const v16bf v = load_a_frag_f32(row, ksl * 32, hf);
        *(v16bf*)(apanel + (slot << 4)) = v;
    }
    __syncthreads();

    const __bf16* bbase = WdP + ((size_t)((wave * 4) * 8 * 32 + lane) << 4);
    const __bf16* abase = apanel + (lane << 4);

    v8f acc[4][2] = {};
    #pragma unroll 2
    for (int ks = 0; ks < 8; ++ks) {
        const v16bf a0 = *(const v16bf*)(abase + (ks * 2 + 0) * (32 * 16));
        const v16bf a1 = *(const v16bf*)(abase + (ks * 2 + 1) * (32 * 16));
        #pragma unroll
        for (int t = 0; t < 4; ++t) {
            const v16bf bf =
                *(const v16bf*)(bbase + (size_t)(t * 8 + ks) * (32 * 16));
            acc[t][0] = __builtin_amdgcn_wmma_f32_16x16x32_bf16(
                false, a0, false, bf, (short)0, acc[t][0], false, false);
            acc[t][1] = __builtin_amdgcn_wmma_f32_16x16x32_bf16(
                false, a1, false, bf, (short)0, acc[t][1], false, false);
        }
    }

    const int ncol = wave * 64 + mrow;
    float bias[4], vv[4];
    #pragma unroll
    for (int t = 0; t < 4; ++t) {
        const int n = ncol + t * 16;
        bias[t] = Dc[(size_t)b * (2 * HDIM) + n];
        vv[t]   = decv[n];
    }
    #pragma unroll
    for (int r = 0; r < 8; ++r) {
        float x0 = 0.0f, x1 = 0.0f;
        #pragma unroll
        for (int t = 0; t < 4; ++t) {
            x0 += tanh_fast(acc[t][0][r] + bias[t]) * vv[t];
            x1 += tanh_fast(acc[t][1][r] + bias[t]) * vv[t];
        }
        x0 += __shfl_xor(x0, 1); x1 += __shfl_xor(x1, 1);
        x0 += __shfl_xor(x0, 2); x1 += __shfl_xor(x1, 2);
        x0 += __shfl_xor(x0, 4); x1 += __shfl_xor(x1, 4);
        x0 += __shfl_xor(x0, 8); x1 += __shfl_xor(x1, 8);
        if (mrow == r) {
            atomicAdd(&sm[r + half * 8], x0);
            atomicAdd(&sm[16 + r + half * 8], x1);
        }
    }
    __syncthreads();
    if (tid < 32 && (s0 + tid) < SDIM)
        scores2[(size_t)b * SDIM + s0 + tid] = sm[tid];
}

// ---------------------------------------------------------------------------
// Kernel 7: final softmax over S -> outputs. One block per b.
// ---------------------------------------------------------------------------
__global__ __launch_bounds__(256) void softmax_out_kernel(
    const float* __restrict__ scores2, float* __restrict__ out)
{
    const int b = blockIdx.x;
    const int tid = threadIdx.x;
    __shared__ float att[SDIM];
    __shared__ float red[256];

    float mx = -3.0e38f;
    for (int s = tid; s < SDIM; s += 256) {
        const float v = scores2[(size_t)b * SDIM + s];
        att[s] = v;
        mx = fmaxf(mx, v);
    }
    red[tid] = mx;
    __syncthreads();
    for (int off = 128; off > 0; off >>= 1) {
        if (tid < off) red[tid] = fmaxf(red[tid], red[tid + off]);
        __syncthreads();
    }
    mx = red[0];
    __syncthreads();

    float sum = 0.0f;
    for (int s = tid; s < SDIM; s += 256) {
        const float e = __expf(att[s] - mx);
        att[s] = e;
        sum += e;
    }
    red[tid] = sum;
    __syncthreads();
    for (int off = 128; off > 0; off >>= 1) {
        if (tid < off) red[tid] += red[tid + off];
        __syncthreads();
    }
    const float inv = 1.0f / red[0];

    for (int s = tid; s < SDIM; s += 256)
        out[(size_t)b * SDIM + s] = att[s] * inv;
}

// ---------------------------------------------------------------------------
// Host-side launcher
// ---------------------------------------------------------------------------
extern "C" void kernel_launch(void* const* d_in, const int* in_sizes, int n_in,
                              void* d_out, int out_size, void* d_ws, size_t ws_size,
                              hipStream_t stream) {
    const float* di    = (const float*)d_in[0];   // [B, 2]
    const float* lh    = (const float*)d_in[1];   // [B, H]
    const float* statc = (const float*)d_in[2];   // [B, S, H]
    const float* dynam = (const float*)d_in[3];   // [B, S, H]
    const float* embW  = (const float*)d_in[4];   // [H, 2]
    const float* embb  = (const float*)d_in[5];   // [H]
    const float* Wih   = (const float*)d_in[6];   // [3H, H]
    const float* Whh   = (const float*)d_in[7];   // [3H, H]
    const float* bih   = (const float*)d_in[8];   // [3H]
    const float* bhh   = (const float*)d_in[9];   // [3H]
    const float* attnW = (const float*)d_in[10];  // [3H, 3H]
    const float* attnv = (const float*)d_in[11];  // [3H]
    const float* decW  = (const float*)d_in[12];  // [2H, 2H]
    const float* decv  = (const float*)d_in[13];  // [2H]

    float* out = (float*)d_out;                   // [B*S] outputs ++ [B*H] hidden

    float*  ws      = (float*)d_ws;
    float*  h_new   = ws;                 // 32768 floats
    float*  Ah      = ws + 32768;         // 98304
    float*  scores  = ws + 131072;        // 128000
    float*  context = ws + 259072;        // 32768
    float*  Dc      = ws + 291840;        // 65536
    float*  scores2 = ws + 357376;        // 128000
    __bf16* WaP     = (__bf16*)(ws + 485376);  // 48*16*32*16 bf16 = 786KB
    __bf16* WdP     = (__bf16*)(ws + 681984);  // 32*8*32*16  bf16 = 262KB
                                               // total ws: ~2.99 MB

    // One-shot weight packing (bf16, B-fragment lane order)
    pack_w_kernel<<<96, 256, 0, stream>>>(attnW, WaP, 3 * HDIM, 16);
    pack_w_kernel<<<32, 256, 0, stream>>>(decW,  WdP, 2 * HDIM, 8);

    gru_kernel<<<BDIM, 256, 0, stream>>>(di, lh, embW, embb, Wih, Whh, bih, bhh,
                                         h_new, out + (size_t)BDIM * SDIM);
    ah_kernel<<<dim3(BDIM, 3), 256, 0, stream>>>(attnW, h_new, Ah);
    attn_scores_kernel<<<BDIM * MT32, 256, 0, stream>>>(statc, dynam, WaP,
                                                        attnv, Ah, scores);
    softmax_context_kernel<<<BDIM, 256, 0, stream>>>(scores, statc, context);
    dc_kernel<<<dim3(BDIM, 2), 256, 0, stream>>>(decW, context, Dc);
    ptr_scores_kernel<<<BDIM * MT32, 256, 0, stream>>>(statc, WdP, decv, Dc,
                                                       scores2);
    softmax_out_kernel<<<BDIM, 256, 0, stream>>>(scores2, out);
}
